// TNet_773094113290
// MI455X (gfx1250) — matrix operand order repaired and checked by reading, not compiled, not run on hardware
//
#include <hip/hip_runtime.h>
#include <hip/hip_bf16.h>
#include <stdint.h>

typedef _Float16 f16;
typedef __attribute__((ext_vector_type(16))) _Float16 v16h;
typedef __attribute__((ext_vector_type(8)))  _Float16 v8h;
typedef __attribute__((ext_vector_type(8)))  float    v8f;

#define EPSBN 1e-5f
#define NB    16      // batch
#define NP    2048    // points
#define KNN   20      // neighbors

struct alignas(16) H8 { _Float16 h[8]; };

// Async copy of one 16B packet from global (saddr base + per-lane offset) into
// LDS at per-lane byte offset. Tracked by ASYNCcnt.
__device__ __forceinline__ void async_g2l_b128(uint32_t lds_off, int global_off,
                                               uint64_t gbase) {
  asm volatile("global_load_async_to_lds_b128 %0, %1, %2"
               :
               : "v"(lds_off), "v"(global_off), "s"(gbase)
               : "memory");
}
__device__ __forceinline__ void wait_async0() {
  asm volatile("s_wait_asynccnt 0x0" ::: "memory");
}

// ---------------------------------------------------------------------------
// Prep: fold BN (+conv bias) into per-channel scale/bias
//   y = s*acc + t,  s = gamma*rsqrt(var+eps), t = s*(bias-mean)+beta
// ---------------------------------------------------------------------------
__global__ void bnfold_kernel(const float* __restrict__ ga, const float* __restrict__ be,
                              const float* __restrict__ mn, const float* __restrict__ vr,
                              const float* __restrict__ bias,
                              float* __restrict__ s, float* __restrict__ t, int c) {
  int i = blockIdx.x * blockDim.x + threadIdx.x;
  if (i < c) {
    float sc = ga[i] * rsqrtf(vr[i] + EPSBN);
    s[i] = sc;
    t[i] = sc * (bias[i] - mn[i]) + be[i];
  }
}

// ---------------------------------------------------------------------------
// Prep: pre-swizzle weights (cout x cin, f32) into CDNA5 WMMA A-fragment
// layout (16x32 f16 per tile/chunk), one contiguous v16h (32B) per lane.
// A order used by this code (contiguous k, B loads match): m = lane&15; element h of lane:
//   k = (lane>=16?16:0) + h
// ---------------------------------------------------------------------------
__global__ void frag_swizzle_kernel(const float* __restrict__ W, f16* __restrict__ out,
                                    int cin, int tiles, int chunks) {
  int total = tiles * chunks * 32 * 16;
  for (int i = blockIdx.x * blockDim.x + threadIdx.x; i < total;
       i += gridDim.x * blockDim.x) {
    int h    = i & 15;
    int lane = (i >> 4) & 31;
    int tc   = i >> 9;              // tile*chunks + chunk
    int chunk = tc % chunks;
    int tile  = tc / chunks;
    int k = chunk * 32 + ((lane >= 16) ? 16 : 0) + h;
    int m = tile * 16 + (lane & 15);
    float val = (k < cin) ? W[m * cin + k] : 0.0f;
    out[i] = (f16)val;
  }
}

__global__ void zerof_kernel(float* __restrict__ p, int n) {
  int i = blockIdx.x * blockDim.x + threadIdx.x;
  if (i < n) p[i] = 0.0f;
}

// ---------------------------------------------------------------------------
// kNN: x[b] staged in LDS via async copy; one thread per point; register
// top-20 (ascending), tie-break lower index first (matches lax.top_k).
// grid = NB*8 blocks of 256 threads
// ---------------------------------------------------------------------------
__global__ void knn_kernel(const float* __restrict__ x, int* __restrict__ knn) {
  __shared__ __attribute__((aligned(16))) float xl[3 * NP];   // 24 KB
  int b   = blockIdx.x >> 3;
  int seg = blockIdx.x & 7;
  const float* xb = x + b * 3 * NP;
  uint32_t shbase = (uint32_t)(size_t)&xl[0];
  for (int i = threadIdx.x; i < (3 * NP) / 4; i += 256)       // 1536 x 16B
    async_g2l_b128(shbase + i * 16, i * 16, (uint64_t)(size_t)xb);
  wait_async0();
  __syncthreads();
  int n = seg * 256 + threadIdx.x;
  float x0 = xl[n], x1 = xl[NP + n], x2 = xl[2 * NP + n];
  float bd[KNN]; int bi[KNN];
#pragma unroll
  for (int j = 0; j < KNN; ++j) { bd[j] = 3.4e38f; bi[j] = 0; }
  for (int m = 0; m < NP; ++m) {
    float dx = x0 - xl[m], dy = x1 - xl[NP + m], dz = x2 - xl[2 * NP + m];
    float d = dx * dx + dy * dy + dz * dz;
    if (d < bd[KNN - 1]) {
      bd[KNN - 1] = d; bi[KNN - 1] = m;
#pragma unroll
      for (int j = KNN - 1; j > 0; --j) {
        if (bd[j - 1] > bd[j]) {
          float td = bd[j - 1]; bd[j - 1] = bd[j]; bd[j] = td;
          int   ti = bi[j - 1]; bi[j - 1] = bi[j]; bi[j] = ti;
        }
      }
    }
  }
  int* o = knn + (b * NP + n) * KNN;
#pragma unroll
  for (int j = 0; j < KNN; ++j) o[j] = bi[j];
}

// ---------------------------------------------------------------------------
// Fused edge conv: h0(6,pad32) -> W1(64x6) WMMA -> BN/ReLU -> W2(128x64) WMMA
// -> BN/ReLU -> max over K=20 -> h2 (b, n, 128) f16.
// Block = (batch b, group of 32 points) => 640 edge columns. 256 thr / 8 waves.
// grid = NB*64 blocks.
// ---------------------------------------------------------------------------
__global__ void edgeconv_kernel(const float* __restrict__ x, const int* __restrict__ knn,
                                const f16* __restrict__ w1f, const f16* __restrict__ w2f,
                                const float* __restrict__ s1, const float* __restrict__ t1,
                                const float* __restrict__ s2, const float* __restrict__ t2,
                                f16* __restrict__ h2) {
  __shared__ __attribute__((aligned(16))) f16 sh0[640][8];   // 10 KB (k>=6 zero)
  __shared__ __attribute__((aligned(32))) f16 sh1[160][64];  // 20 KB, [col][ch]
  __shared__ int smax[128][32];                              // 16 KB float-bits
  int b    = blockIdx.x >> 6;
  int grp  = blockIdx.x & 63;
  int base = grp * 32;
  int tid = threadIdx.x, lane = tid & 31, wave = tid >> 5;
  const float* xb = x + b * 3 * NP;

  for (int i = tid; i < 128 * 32; i += 256) ((int*)smax)[i] = 0;
  for (int colc = tid; colc < 640; colc += 256) {
    int p = colc / 20, kk = colc - p * 20;
    int n = base + p;
    int nb = knn[(b * NP + n) * KNN + kk];
#pragma unroll
    for (int c = 0; c < 3; ++c) {
      float cen = xb[c * NP + n];
      float nv  = xb[c * NP + nb];
      sh0[colc][c]     = (f16)cen;
      sh0[colc][c + 3] = (f16)(nv - cen);
    }
    sh0[colc][6] = (f16)0.f;
    sh0[colc][7] = (f16)0.f;
  }
  __syncthreads();

  for (int q = 0; q < 4; ++q) {           // 4 quarters of 160 columns
    // ---- layer 1: 4 mtiles x 10 ntiles ----
    for (int job = wave; job < 40; job += 8) {
      int mtile = job / 10, ntile = job - mtile * 10;
      int lc  = ntile * 16 + (lane & 15);
      int col = q * 160 + lc;
      v16h a = *(const v16h*)&w1f[(mtile * 32 + lane) * 16];
      v16h bf = {};
      if (lane < 16) {                    // k = h (0..15); k>=8 already zero
        v8h lo = *(const v8h*)&sh0[col][0];
#pragma unroll
        for (int i2 = 0; i2 < 8; ++i2) bf[i2] = lo[i2];
      }
      v8f c = {};
      c = __builtin_amdgcn_wmma_f32_16x16x32_f16(false, a, false, bf, (short)0, c, false, false);
      int mbase = mtile * 16 + ((lane >= 16) ? 8 : 0);
      H8 pk;
#pragma unroll
      for (int v = 0; v < 8; ++v) {
        int m = mbase + v;
        float r = s1[m] * c[v] + t1[m];
        pk.h[v] = (f16)(r > 0.f ? r : 0.f);
      }
      *(H8*)&sh1[lc][mbase] = pk;         // ds_store_b128
    }
    __syncthreads();
    // ---- layer 2: 8 mtiles x 10 ntiles, K=64 in 2 chunks ----
    for (int job = wave; job < 80; job += 8) {
      int mtile = job / 10, ntile = job - mtile * 10;
      int lc = ntile * 16 + (lane & 15);
      v8f c = {};
#pragma unroll
      for (int ch = 0; ch < 2; ++ch) {
        v16h a  = *(const v16h*)&w2f[((mtile * 2 + ch) * 32 + lane) * 16];
        v16h bf = *(const v16h*)&sh1[lc][ch * 32 + ((lane >= 16) ? 16 : 0)];
        c = __builtin_amdgcn_wmma_f32_16x16x32_f16(false, a, false, bf, (short)0, c, false, false);
      }
      int gcol  = q * 160 + lc;
      int p     = gcol / 20;
      int mbase = mtile * 16 + ((lane >= 16) ? 8 : 0);
#pragma unroll
      for (int v = 0; v < 8; ++v) {
        int m = mbase + v;
        float r = s2[m] * c[v] + t2[m];
        r = r > 0.f ? r : 0.f;
        atomicMax(&smax[m][p], __float_as_int(r));  // valid: r >= 0
      }
    }
    __syncthreads();
  }
  for (int i = tid; i < 128 * 32; i += 256) {
    int m = i >> 5, p = i & 31;
    h2[((size_t)(b * NP) + base + p) * 128 + m] = (f16)__int_as_float(smax[m][p]);
  }
}

// ---------------------------------------------------------------------------
// mlp_local 128->1024 (WMMA, K=128 in 4 chunks) + BN/ReLU + global max over N.
// Block = (b, 128-column chunk): stages the 32KB h2 tile into LDS with async
// copies, then all 64 m-tiles (8 per wave) consume it.
// grid = NB * 16 blocks of 256 threads.
// ---------------------------------------------------------------------------
__global__ void mlplocal_kernel(const f16* __restrict__ h2, const f16* __restrict__ w3f,
                                const float* __restrict__ s3, const float* __restrict__ t3,
                                float* __restrict__ g) {
  __shared__ __attribute__((aligned(32))) f16 sh[128 * 128];  // 32 KB, [ncol][ch]
  int b      = blockIdx.x >> 4;
  int nchunk = blockIdx.x & 15;
  int tid = threadIdx.x, lane = tid & 31, wave = tid >> 5;
  const f16* gptr = h2 + (size_t)(b * NP + nchunk * 128) * 128;  // contiguous 32KB
  uint32_t shbase = (uint32_t)(size_t)&sh[0];
  for (int i = tid; i < 2048; i += 256)                          // 2048 x 16B
    async_g2l_b128(shbase + i * 16, i * 16, (uint64_t)(size_t)gptr);
  wait_async0();
  __syncthreads();

  for (int mi = 0; mi < 8; ++mi) {
    int mt = wave * 8 + mi;                 // 0..63
    v8f mx = {};
    for (int ntile = 0; ntile < 8; ++ntile) {
      int nl = ntile * 16 + (lane & 15);
      const f16* col = &sh[nl * 128];
      v8f c = {};
#pragma unroll
      for (int ch = 0; ch < 4; ++ch) {
        v16h a  = *(const v16h*)&w3f[((mt * 4 + ch) * 32 + lane) * 16];
        v16h bf = *(const v16h*)&col[ch * 32 + ((lane >= 16) ? 16 : 0)];
        c = __builtin_amdgcn_wmma_f32_16x16x32_f16(false, a, false, bf, (short)0, c, false, false);
      }
      int mbase = mt * 16 + ((lane >= 16) ? 8 : 0);
#pragma unroll
      for (int v = 0; v < 8; ++v) {
        float r = s3[mbase + v] * c[v] + t3[mbase + v];
        r = r > 0.f ? r : 0.f;
        mx[v] = mx[v] > r ? mx[v] : r;
      }
    }
    int mbase = mt * 16 + ((lane >= 16) ? 8 : 0);
#pragma unroll
    for (int v = 0; v < 8; ++v)
      atomicMax((int*)&g[b * 1024 + mbase + v], __float_as_int(mx[v]));
  }
}

// ---------------------------------------------------------------------------
// Tail: tiny dense layers (scalar)
// ---------------------------------------------------------------------------
__global__ void dense_relu_kernel(const float* __restrict__ in, const float* __restrict__ W,
                                  const float* __restrict__ s, const float* __restrict__ t,
                                  float* __restrict__ out, int cin, int cout) {
  int i = blockIdx.x * blockDim.x + threadIdx.x;
  if (i >= NB * cout) return;
  int b = i / cout, o = i - b * cout;
  const float* gi = in + b * cin;
  const float* wr = W + o * cin;
  float acc = 0.f;
  for (int k = 0; k < cin; ++k) acc += gi[k] * wr[k];
  float r = s[o] * acc + t[o];
  out[i] = r > 0.f ? r : 0.f;
}

__global__ void final_kernel(const float* __restrict__ in, const float* __restrict__ W,
                             const float* __restrict__ bias, float* __restrict__ out) {
  int i = threadIdx.x;
  if (i >= NB * 9) return;
  int b = i / 9, o = i - b * 9;
  const float* gi = in + b * 256;
  const float* wr = W + o * 256;
  float acc = bias[o];
  for (int k = 0; k < 256; ++k) acc += gi[k] * wr[k];
  if (o == 0 || o == 4 || o == 8) acc += 1.f;   // + eye(3)
  out[i] = acc;
}

// ---------------------------------------------------------------------------
// Launch. Input order (setup_inputs dict insertion order, leaves in nested
// insertion order):
//  0:x 1:ec0.W 2:ec0.b 3-6:ec0.{gamma,beta,mean,var}
//  7:ec1.W 8:ec1.b 9-12:ec1.bn
//  13:ml0.W 14:ml0.b 15-18:ml0.bn
//  19:mg0.W 20:mg0.b 21-24:mg0.bn
//  25:mg1.W 26:mg1.b 27-30:mg1.bn
//  31:lin.W 32:lin.b
// ---------------------------------------------------------------------------
extern "C" void kernel_launch(void* const* d_in, const int* in_sizes, int n_in,
                              void* d_out, int out_size, void* d_ws, size_t ws_size,
                              hipStream_t stream) {
  (void)in_sizes; (void)n_in; (void)out_size; (void)ws_size;
  const float* x   = (const float*)d_in[0];
  const float* W1  = (const float*)d_in[1];
  const float* W2  = (const float*)d_in[7];
  const float* W3  = (const float*)d_in[13];
  const float* W4  = (const float*)d_in[19];
  const float* W5  = (const float*)d_in[25];
  const float* WL  = (const float*)d_in[31];
  const float* bL  = (const float*)d_in[32];
  float* out = (float*)d_out;

  uint8_t* ws = (uint8_t*)d_ws;
  size_t off = 0;
  auto alloc = [&](size_t bytes) -> uint8_t* {
    uint8_t* p = ws + off;
    off = (off + bytes + 255) & ~(size_t)255;
    return p;
  };
  int*   knn  = (int*)  alloc((size_t)NB * NP * KNN * 4);     // 2.6 MB
  f16*   h2   = (f16*)  alloc((size_t)NB * NP * 128 * 2);     // 8 MB
  f16*   w1f  = (f16*)  alloc(4  * 1 * 32 * 16 * 2);
  f16*   w2f  = (f16*)  alloc(8  * 2 * 32 * 16 * 2);
  f16*   w3f  = (f16*)  alloc(64 * 4 * 32 * 16 * 2);
  float* s1 = (float*)alloc(64 * 4);   float* t1 = (float*)alloc(64 * 4);
  float* s2 = (float*)alloc(128 * 4);  float* t2 = (float*)alloc(128 * 4);
  float* s3 = (float*)alloc(1024 * 4); float* t3 = (float*)alloc(1024 * 4);
  float* s4 = (float*)alloc(512 * 4);  float* t4 = (float*)alloc(512 * 4);
  float* s5 = (float*)alloc(256 * 4);  float* t5 = (float*)alloc(256 * 4);
  float* g  = (float*)alloc(NB * 1024 * 4);
  float* g2 = (float*)alloc(NB * 512 * 4);
  float* g3 = (float*)alloc(NB * 256 * 4);

  // ---- prep ----
  frag_swizzle_kernel<<<8,   256, 0, stream>>>(W1, w1f, 6,   4,  1);
  frag_swizzle_kernel<<<32,  256, 0, stream>>>(W2, w2f, 64,  8,  2);
  frag_swizzle_kernel<<<512, 256, 0, stream>>>(W3, w3f, 128, 64, 4);
  bnfold_kernel<<<1, 256, 0, stream>>>((const float*)d_in[3],  (const float*)d_in[4],
                                       (const float*)d_in[5],  (const float*)d_in[6],
                                       (const float*)d_in[2],  s1, t1, 64);
  bnfold_kernel<<<1, 256, 0, stream>>>((const float*)d_in[9],  (const float*)d_in[10],
                                       (const float*)d_in[11], (const float*)d_in[12],
                                       (const float*)d_in[8],  s2, t2, 128);
  bnfold_kernel<<<4, 256, 0, stream>>>((const float*)d_in[15], (const float*)d_in[16],
                                       (const float*)d_in[17], (const float*)d_in[18],
                                       (const float*)d_in[14], s3, t3, 1024);
  bnfold_kernel<<<2, 256, 0, stream>>>((const float*)d_in[21], (const float*)d_in[22],
                                       (const float*)d_in[23], (const float*)d_in[24],
                                       (const float*)d_in[20], s4, t4, 512);
  bnfold_kernel<<<1, 256, 0, stream>>>((const float*)d_in[27], (const float*)d_in[28],
                                       (const float*)d_in[29], (const float*)d_in[30],
                                       (const float*)d_in[26], s5, t5, 256);
  zerof_kernel<<<64, 256, 0, stream>>>(g, NB * 1024);

  // ---- main pipeline ----
  knn_kernel<<<NB * 8, 256, 0, stream>>>(x, knn);
  edgeconv_kernel<<<NB * 64, 256, 0, stream>>>(x, knn, w1f, w2f, s1, t1, s2, t2, h2);
  mlplocal_kernel<<<NB * 16, 256, 0, stream>>>(h2, w3f, s3, t3, g);
  dense_relu_kernel<<<(NB * 512 + 255) / 256, 256, 0, stream>>>(g,  W4, s4, t4, g2, 1024, 512);
  dense_relu_kernel<<<(NB * 256 + 255) / 256, 256, 0, stream>>>(g2, W5, s5, t5, g3, 512, 256);
  final_kernel<<<1, 256, 0, stream>>>(g3, WL, bL, out);
}
